// Attention_10608569221223
// MI455X (gfx1250) — matrix operand order repaired and checked by reading, not compiled, not run on hardware
//
#include <hip/hip_runtime.h>
#include <hip/hip_bf16.h>
#include <stdint.h>

typedef __attribute__((ext_vector_type(16))) __bf16 v16bf;
typedef __attribute__((ext_vector_type(8)))  float  v8f;
typedef __attribute__((ext_vector_type(4)))  unsigned int u32x4;
typedef __attribute__((ext_vector_type(8)))  int i32x8;
typedef __attribute__((ext_vector_type(4)))  int i32x4;

#define DIM   1024
#define NHEAD 16
#define HD    64
#define BATCH 8
#define SEQ   1024
#define ROWS  (BATCH*SEQ)   // 8192
#define ATT_SCALE 0.125f    // 64^-0.5

#if defined(__has_builtin)
#if __has_builtin(__builtin_amdgcn_tensor_load_to_lds)
#define HAVE_TDM 1
#endif
#endif

union FragU { v16bf v; uint4 q[2]; uint32_t u[8]; };

// A-fragment (16-bit A, 16x32, MxK): lane m = L&15; elems 0..15 = K[koff..koff+15]
// contiguous, koff = (L>>4)*16. Row-major [rows, ld] bf16.
__device__ __forceinline__ v16bf ldA(const __bf16* base, int ld, int rowBase, int k0, int lane) {
  const __bf16* p = base + (size_t)(rowBase + (lane & 15)) * ld + k0 + ((lane >> 4) << 4);
  FragU f;
  f.q[0] = *(const uint4*)(p);
  f.q[1] = *(const uint4*)(p + 8);
  return f.v;
}

// B-fragment (16-bit B, 32x16, KxN) from TRANSPOSED storage [N, K] row-major:
// lane n = L&15; elems 0..15 = K[kb..kb+15] contiguous, kb = (L>>4)*16.
__device__ __forceinline__ v16bf ldB(const __bf16* baseT, int ld, int colBase, int k0, int lane) {
  const __bf16* p = baseT + (size_t)(colBase + (lane & 15)) * ld + k0 + ((lane >> 4) << 4);
  FragU f;
  f.q[0] = *(const uint4*)(p);
  f.q[1] = *(const uint4*)(p + 8);
  return f.v;
}

__device__ __forceinline__ v8f wmma_bf16(v16bf a, v16bf b, v8f c) {
  return __builtin_amdgcn_wmma_f32_16x16x32_bf16(false, a, false, b, (short)0, c, false, false);
}

__device__ __forceinline__ uint32_t pk_bf16x2(float lo, float hi) {
  __bf16 l = (__bf16)lo, h = (__bf16)hi;
  uint16_t lb = __builtin_bit_cast(uint16_t, l);
  uint16_t hb = __builtin_bit_cast(uint16_t, h);
  return ((uint32_t)hb << 16) | (uint32_t)lb;
}

#ifdef HAVE_TDM
// Issue one TDM 2D tile load: tile_y rows x tile_x bf16 elements from row-major
// global tensor (row stride = stride_elems) into contiguous LDS at lds_off.
// D# layout per CDNA5 ISA ch.8: group0 = {ctl, lds_addr, gaddr_lo, gaddr_hi|type},
// group1 = {mask/data_size, dims, tile dims, dim0 stride}. Groups 2/3 unused (2D).
__device__ __forceinline__ void tdm_load_2d(const void* gaddr, unsigned lds_off,
                                            unsigned tile_x, unsigned tile_y,
                                            unsigned stride_elems,
                                            unsigned tdim0, unsigned tdim1) {
  unsigned long long ga = (unsigned long long)(size_t)gaddr;
  u32x4 g0;
  g0[0] = 1u;                                                  // count=1, user D#
  g0[1] = lds_off;                                             // lds_addr
  g0[2] = (unsigned)ga;                                        // global_addr[31:0]
  g0[3] = (unsigned)((ga >> 32) & 0x01FFFFFFu) | (2u << 30);   // addr[56:32], type=2
  i32x8 g1;
  g1[0] = (int)(1u << 16);                                     // data_size=1 (2 bytes)
  g1[1] = (int)((tdim0 & 0xFFFFu) << 16);                      // tensor_dim0[15:0]
  g1[2] = (int)((tdim0 >> 16) | ((tdim1 & 0xFFFFu) << 16));    // td0[31:16], td1[15:0]
  g1[3] = (int)((tdim1 >> 16) | (tile_x << 16));               // td1[31:16], tile_dim0
  g1[4] = (int)(tile_y & 0xFFFFu);                             // tile_dim1, tile_dim2=0
  g1[5] = (int)stride_elems;                                   // tensor_dim0_stride lo
  g1[6] = 0;
  g1[7] = 0;
  i32x4 z4 = {0, 0, 0, 0};
#if defined(__clang_major__) && (__clang_major__ >= 23)
  i32x8 z8 = {0, 0, 0, 0, 0, 0, 0, 0};
  __builtin_amdgcn_tensor_load_to_lds(g0, g1, z4, z4, z8, 0);
#else
  __builtin_amdgcn_tensor_load_to_lds(g0, g1, z4, z4, 0);
#endif
}
#endif

// ---------------- K0: fp32 -> bf16 converts ----------------
__global__ void k_cvt(const float* __restrict__ src, __bf16* __restrict__ dst, int n) {
  int i = blockIdx.x * blockDim.x + threadIdx.x;
  if (i < n) dst[i] = (__bf16)src[i];
}

// W [K, Ncols] fp32 -> WT [Ncols, K] bf16
__global__ void k_cvt_t(const float* __restrict__ src, __bf16* __restrict__ dstT, int K, int Ncols) {
  int i = blockIdx.x * blockDim.x + threadIdx.x;
  if (i < K * Ncols) {
    int k = i / Ncols, n = i - k * Ncols;
    dstT[(size_t)n * K + k] = (__bf16)src[i];
  }
}

// ---------------- K1: qkv = x @ W_qkv, permuted epilogue ----------------
__global__ void k_qkv(const __bf16* __restrict__ xb, const __bf16* __restrict__ WT,
                      __bf16* __restrict__ Qb, __bf16* __restrict__ Kb, __bf16* __restrict__ VTb) {
  const int lane = threadIdx.x & 31;
  const int gw = blockIdx.x * (blockDim.x >> 5) + (threadIdx.x >> 5);
  const int mt = gw / 48;          // 512 M-tiles of 16 rows
  const int ng = gw - mt * 48;     // 48 N-groups of 64 cols
  v8f acc0 = {}, acc1 = {}, acc2 = {}, acc3 = {};
  for (int kk = 0; kk < DIM; kk += 32) {
    v16bf a = ldA(xb, DIM, mt * 16, kk, lane);
    acc0 = wmma_bf16(a, ldB(WT, DIM, ng * 64 +  0, kk, lane), acc0);
    acc1 = wmma_bf16(a, ldB(WT, DIM, ng * 64 + 16, kk, lane), acc1);
    acc2 = wmma_bf16(a, ldB(WT, DIM, ng * 64 + 32, kk, lane), acc2);
    acc3 = wmma_bf16(a, ldB(WT, DIM, ng * 64 + 48, kk, lane), acc3);
  }
  v8f accs[4] = {acc0, acc1, acc2, acc3};
  #pragma unroll
  for (int t = 0; t < 4; t++) {
    #pragma unroll
    for (int i = 0; i < 8; i++) {
      int row = mt * 16 + i + ((lane >> 4) << 3);       // global row in [0,8192)
      int col = ng * 64 + t * 16 + (lane & 15);         // global col in [0,3072)
      int which = col >> 10;                            // 0=q 1=k 2=v
      int rem = col & 1023;
      int h = rem >> 6, d = rem & 63;
      int bi = row >> 10, n = row & 1023;
      size_t bh = (size_t)(bi * NHEAD + h);
      __bf16 bv = (__bf16)accs[t][i];
      if (which == 0)      Qb[(bh << 16) + ((size_t)n << 6) + d] = bv;
      else if (which == 1) Kb[(bh << 16) + ((size_t)n << 6) + d] = bv;
      else                 VTb[(bh << 16) + ((size_t)d << 10) + n] = bv;
    }
  }
}

// ---------------- K2: flash attention (transposed form, TDM-staged K/V in LDS) ----
// Block = 4 waves, all on the same (b,h), consecutive query tiles: the K-tile and
// V^T-tile for each 32-key step are staged once into LDS (TDM) and shared 4x.
// S^T = K Q^T keeps queries on the lane axis -> softmax stats are per-lane scalars.
__global__ void k_attn(const __bf16* __restrict__ Qb, const __bf16* __restrict__ Kb,
                       const __bf16* __restrict__ VTb, __bf16* __restrict__ AO) {
  __shared__ __align__(16) __bf16 shK[32 * HD];    // [key 0..31][d 0..63]  4 KB
  __shared__ __align__(16) __bf16 shVT[HD * 32];   // [d 0..63][key 0..31]  4 KB

  const int lane = threadIdx.x & 31;
  const int wid = threadIdx.x >> 5;
  const int bh = blockIdx.x >> 4;                 // (b*16 + h)
  const int qt = ((blockIdx.x & 15) << 2) + wid;  // query tile 0..63
  const __bf16* Qp = Qb + ((size_t)bh << 16);     // [1024, 64]
  const __bf16* Kp = Kb + ((size_t)bh << 16);     // [1024, 64]
  const __bf16* Vp = VTb + ((size_t)bh << 16);    // [64, 1024]

  // Q^T as B-fragments (d x queries), reused across the whole key loop
  v16bf qb0 = ldB(Qp, HD, qt * 16, 0, lane);
  v16bf qb1 = ldB(Qp, HD, qt * 16, 32, lane);

  v8f o0 = {}, o1 = {}, o2 = {}, o3 = {};
  float m = -1e30f, l = 0.f;
  const bool hi = (lane >= 16);

  for (int j = 0; j < SEQ; j += 32) {
#ifdef HAVE_TDM
    if (wid == 0) {
      // K tile: 32 rows x 64 cols, contiguous rows (stride 64)
      tdm_load_2d(Kp + (size_t)j * HD, (unsigned)(size_t)shK, HD, 32, HD, HD, SEQ);
      // V^T tile: 64 rows x 32 cols at col offset j (stride 1024)
      tdm_load_2d(Vp + j, (unsigned)(size_t)shVT, 32, HD, SEQ, SEQ, HD);
      __builtin_amdgcn_s_wait_tensorcnt(0);
    }
    __syncthreads();
#else
    {
      const int tid = threadIdx.x;
      const uint4* srcK = (const uint4*)(Kp + (size_t)j * HD);  // contiguous 4 KB
      for (int i = tid; i < 256; i += 128) ((uint4*)shK)[i] = srcK[i];
      for (int i = tid; i < 256; i += 128) {
        int r = i >> 2, c = (i & 3) << 3;
        ((uint4*)shVT)[i] = *(const uint4*)(Vp + (size_t)r * SEQ + j + c);
      }
    }
    __syncthreads();
#endif

    v8f z0 = {}, z1 = {};
    // S^T tiles for keys j..j+15 and j+16..j+31 (from LDS)
    v8f s0 = wmma_bf16(ldA(shK, HD, 0,  32, lane), qb1,
             wmma_bf16(ldA(shK, HD, 0,   0, lane), qb0, z0));
    v8f s1 = wmma_bf16(ldA(shK, HD, 16, 32, lane), qb1,
             wmma_bf16(ldA(shK, HD, 16,  0, lane), qb0, z1));

    float tmax = -1e30f;
    #pragma unroll
    for (int i = 0; i < 8; i++) {
      s0[i] *= ATT_SCALE; s1[i] *= ATT_SCALE;
      tmax = fmaxf(tmax, fmaxf(s0[i], s1[i]));
    }
    tmax = fmaxf(tmax, __shfl_xor(tmax, 16, 32));
    float mnew = fmaxf(m, tmax);
    float alpha = __expf(m - mnew);
    m = mnew;

    float p0[8], p1[8];
    float tsum = 0.f;
    #pragma unroll
    for (int i = 0; i < 8; i++) {
      p0[i] = __expf(s0[i] - mnew);
      p1[i] = __expf(s1[i] - mnew);
      tsum += p0[i] + p1[i];
    }
    tsum += __shfl_xor(tsum, 16, 32);
    l = l * alpha + tsum;

    #pragma unroll
    for (int i = 0; i < 8; i++) { o0[i] *= alpha; o1[i] *= alpha; o2[i] *= alpha; o3[i] *= alpha; }

    // P^T C-layout -> B-fragment (32 keys x 16 queries): one xor16 shuffle per tile.
    float sel[16];
    #pragma unroll
    for (int i = 0; i < 8; i++) {
      float q0 = __shfl_xor(p0[i], 16, 32);
      float q1 = __shfl_xor(p1[i], 16, 32);
      sel[i]     = hi ? q1 : p0[i];
      sel[8 + i] = hi ? p1[i] : q0;
    }
    FragU pb;
    #pragma unroll
    for (int jj = 0; jj < 8; jj++) pb.u[jj] = pk_bf16x2(sel[2 * jj], sel[2 * jj + 1]);

    // O^T += V^T(d-tile, keys 0..31 local) * P^T   (V^T from LDS, ld=32)
    o0 = wmma_bf16(ldA(shVT, 32,  0, 0, lane), pb.v, o0);
    o1 = wmma_bf16(ldA(shVT, 32, 16, 0, lane), pb.v, o1);
    o2 = wmma_bf16(ldA(shVT, 32, 32, 0, lane), pb.v, o2);
    o3 = wmma_bf16(ldA(shVT, 32, 48, 0, lane), pb.v, o3);

    __syncthreads();   // protect LDS tiles before next TDM overwrite
  }

  float linv = 1.f / l;   // per query (= per lane)
  int b = bh >> 4, h = bh & 15;
  int n = qt * 16 + (lane & 15);
  __bf16* dst = AO + ((size_t)b * SEQ + n) * DIM + h * HD;
  v8f os[4] = {o0, o1, o2, o3};
  #pragma unroll
  for (int t = 0; t < 4; t++) {
    #pragma unroll
    for (int i = 0; i < 8; i++) {
      int d = t * 16 + i + ((lane >> 4) << 3);
      dst[d] = (__bf16)(os[t][i] * linv);
    }
  }
}

// ---------------- K3: out = attn_out @ W_proj + b_proj (fp32 out) ----------------
__global__ void k_proj(const __bf16* __restrict__ A, const __bf16* __restrict__ WT,
                       const float* __restrict__ bias, float* __restrict__ out) {
  const int lane = threadIdx.x & 31;
  const int gw = blockIdx.x * (blockDim.x >> 5) + (threadIdx.x >> 5);
  const int mt = gw >> 4;        // 512 M-tiles
  const int ng = gw & 15;        // 16 N-groups of 64
  v8f acc0 = {}, acc1 = {}, acc2 = {}, acc3 = {};
  for (int kk = 0; kk < DIM; kk += 32) {
    v16bf a = ldA(A, DIM, mt * 16, kk, lane);
    acc0 = wmma_bf16(a, ldB(WT, DIM, ng * 64 +  0, kk, lane), acc0);
    acc1 = wmma_bf16(a, ldB(WT, DIM, ng * 64 + 16, kk, lane), acc1);
    acc2 = wmma_bf16(a, ldB(WT, DIM, ng * 64 + 32, kk, lane), acc2);
    acc3 = wmma_bf16(a, ldB(WT, DIM, ng * 64 + 48, kk, lane), acc3);
  }
  v8f accs[4] = {acc0, acc1, acc2, acc3};
  #pragma unroll
  for (int t = 0; t < 4; t++) {
    #pragma unroll
    for (int i = 0; i < 8; i++) {
      int row = mt * 16 + i + ((lane >> 4) << 3);
      int col = ng * 64 + t * 16 + (lane & 15);
      out[(size_t)row * DIM + col] = accs[t][i] + bias[col];
    }
  }
}

extern "C" void kernel_launch(void* const* d_in, const int* in_sizes, int n_in,
                              void* d_out, int out_size, void* d_ws, size_t ws_size,
                              hipStream_t stream) {
  (void)in_sizes; (void)n_in; (void)out_size; (void)ws_size;
  const float* x     = (const float*)d_in[0];
  const float* Wqkv  = (const float*)d_in[1];
  const float* Wproj = (const float*)d_in[2];
  const float* bproj = (const float*)d_in[3];
  float* out = (float*)d_out;

  char* ws = (char*)d_ws;
  __bf16* xb     = (__bf16*)(ws + 0);             // 16 MiB  [8192,1024]
  __bf16* WqkvT  = (__bf16*)(ws + (16u << 20));   // 6 MiB   [3072,1024]
  __bf16* WprojT = (__bf16*)(ws + (22u << 20));   // 2 MiB   [1024,1024]
  __bf16* Qb     = (__bf16*)(ws + (24u << 20));   // 16 MiB  [128,1024,64]
  __bf16* Kb     = (__bf16*)(ws + (40u << 20));   // 16 MiB  [128,1024,64]
  __bf16* VTb    = (__bf16*)(ws + (56u << 20));   // 16 MiB  [128,64,1024]
  __bf16* AO     = (__bf16*)(ws + (72u << 20));   // 16 MiB  [8192,1024]

  k_cvt  <<<(ROWS * DIM + 255) / 256, 256, 0, stream>>>(x, xb, ROWS * DIM);
  k_cvt_t<<<(DIM * 3 * DIM + 255) / 256, 256, 0, stream>>>(Wqkv, WqkvT, DIM, 3 * DIM);
  k_cvt_t<<<(DIM * DIM + 255) / 256, 256, 0, stream>>>(Wproj, WprojT, DIM, DIM);

  // 512 M-tiles * 48 N-groups = 24576 waves / 4 waves per block
  k_qkv <<<6144, 128, 0, stream>>>(xb, WqkvT, Qb, Kb, VTb);
  // 128 (b,h) * 16 blocks of 4 query-tile waves
  k_attn<<<2048, 128, 0, stream>>>(Qb, Kb, VTb, AO);
  // 512 * 16 = 8192 waves
  k_proj<<<2048, 128, 0, stream>>>(AO, WprojT, bproj, out);
}
